// WindowAttention4D_32152125178540
// MI455X (gfx1250) — compile-verified, matrix operand order checked by reading, not run on hardware
//
#include <hip/hip_runtime.h>

// ---------------------------------------------------------------------------
// Fused Swin window attention for gfx1250 (MI455X), wave32 + WMMA f16.
//   x:[4096,128,128] f32, mask:[64,128,128] f32, w_qkv:[384,128] f32,
//   w_proj:[128,128] f32, b_proj:[128] f32 -> out:[4096,128,128] f32
// One workgroup (8 wave32) per window; intermediates in 128 KB LDS.
// All GEMMs are computed transposed (D = [feature, token]) so the per-lane
// D-fragment values are contiguous -> b128 stores, in-register softmax rows,
// and in-lane P re-fragmentation (no LDS staging between S and P@V).
// ---------------------------------------------------------------------------

typedef _Float16 h16;
typedef __attribute__((ext_vector_type(16))) _Float16 v16h;
typedef __attribute__((ext_vector_type(8)))  float    v8f;
typedef __attribute__((ext_vector_type(4)))  int      v4i;

#define NTOK   128
#define NCH    128
#define NHEAD  4
#define HD     32    // head dim == one 16x16x32 WMMA K step
#define NWMASK 64
#define QSCALE 0.17677669529663687f   // 32^-0.5

#if defined(__gfx1250__) &&                                          \
    __has_builtin(__builtin_amdgcn_global_load_async_to_lds_b128) && \
    __has_builtin(__builtin_amdgcn_s_wait_asynccnt)
#define HAVE_ASYNC_LDS 1
#else
#define HAVE_ASYNC_LDS 0
#endif

#define AS1 __attribute__((address_space(1)))
#define AS3 __attribute__((address_space(3)))

// Fragment from a row-major f16 matrix (ISA 7.12.2: lane-group g covers
// K in {8g..8g+7, 8g+16..8g+23}).  As A: A(row0+lane%16, col0+kmap(e,g)).
// As B (transposed source): B(k=col0+kmap(e,g), n=row0+lane%16) = src[n][k].
__device__ __forceinline__ v16h load_frag_h(const h16* __restrict__ src,
                                            int row0, int col0, int ld, int lane) {
  const int r = lane & 15, g = lane >> 4;
  const h16* p = src + (row0 + r) * ld + col0 + 8 * g;
  v16h a;
#pragma unroll
  for (int e = 0; e < 8; ++e) { a[e] = p[e]; a[e + 8] = p[e + 16]; }
  return a;
}

// Same fragment from a row-major f32 (global, L2-resident weights), converting.
__device__ __forceinline__ v16h load_frag_f(const float* __restrict__ src,
                                            int row0, int col0, int ld, int lane) {
  const int r = lane & 15, g = lane >> 4;
  const float* p = src + (row0 + r) * ld + col0 + 8 * g;
  v16h a;
#pragma unroll
  for (int e = 0; e < 8; ++e) { a[e] = (h16)p[e]; a[e + 8] = (h16)p[e + 16]; }
  return a;
}

__device__ __forceinline__ v8f wmma_f16(v16h a, v16h b, v8f c) {
  return __builtin_amdgcn_wmma_f32_16x16x32_f16(false, a, false, b,
                                                (short)0, c, false, false);
}

__global__ void __launch_bounds__(256)
swin_window_attention_kernel(const float* __restrict__ x,
                             const float* __restrict__ mask,
                             const float* __restrict__ w_qkv,
                             const float* __restrict__ w_proj,
                             const float* __restrict__ b_proj,
                             float* __restrict__ out) {
  // 128 KB LDS, four 32 KB planes with stage-based aliases.
  __shared__ h16 smem[4 * NTOK * NCH];
  h16* xs = smem;                  // x in f16 (dead after stage A)
  h16* qs = smem + 1 * NTOK * NCH; // scaled Q [tok][c]   (later: attn-out)
  h16* ks = smem + 2 * NTOK * NCH; // K [tok][c]          (startup: f32 staging lo)
  h16* vt = smem + 3 * NTOK * NCH; // V^T [c][tok]        (startup: f32 staging hi)

  const int b    = blockIdx.x;
  const int tid  = threadIdx.x;
  const int lane = tid & 31;
  const int wave = __builtin_amdgcn_readfirstlane(tid >> 5);   // SGPR wave id
  const int win  = b & (NWMASK - 1);
  const int g    = lane >> 4;     // D-frag: values m = 8g+r, fixed n = lane%16
  const int nl   = lane & 15;

  const float* xb = x + (size_t)b * NTOK * NCH;

  // ---- Stage 0: window of x -> LDS f16 -----------------------------------
#if HAVE_ASYNC_LDS
  {
    float* stagef = reinterpret_cast<float*>(ks);   // 64 KB staging in K/V planes
#pragma unroll
    for (int it = 0; it < 16; ++it) {
      const int fidx = (it * 256 + tid) * 4;        // 16 B per lane per op
      __builtin_amdgcn_global_load_async_to_lds_b128(
          (AS1 v4i*)(xb + fidx), (AS3 v4i*)(stagef + fidx), 0, 0);
    }
    __builtin_amdgcn_s_wait_asynccnt(0);
    __syncthreads();
#pragma unroll
    for (int it = 0; it < 16; ++it) {
      const int fidx = (it * 256 + tid) * 4;
      const float4 v4 = *reinterpret_cast<const float4*>(stagef + fidx);
      xs[fidx + 0] = (h16)v4.x;  xs[fidx + 1] = (h16)v4.y;
      xs[fidx + 2] = (h16)v4.z;  xs[fidx + 3] = (h16)v4.w;
    }
  }
#else
#pragma unroll
  for (int it = 0; it < 16; ++it) {
    const int fidx = (it * 256 + tid) * 4;
    const float4 v4 = *reinterpret_cast<const float4*>(xb + fidx);
    xs[fidx + 0] = (h16)v4.x;  xs[fidx + 1] = (h16)v4.y;
    xs[fidx + 2] = (h16)v4.z;  xs[fidx + 3] = (h16)v4.w;
  }
#endif
  __syncthreads();

  // ---- Stage A: qkv^T = w_qkv @ x^T  -------------------------------------
  // A = w_qkv rows (features j), B = x^T.  D: m = j, n = token -> Q/K stores
  // are 8 contiguous halves per lane (b128).  24 j-tiles; each wave owns 3.
  for (int t = 0; t < 3; ++t) {
    const int ntile = wave * 3 + t;        // SGPR
    const int j0    = ntile * 16;
    const int kind  = ntile >> 3;          // 0=Q, 1=K, 2=V (SGPR branch)
    v16h aw[4];
#pragma unroll
    for (int kt = 0; kt < 4; ++kt)
      aw[kt] = load_frag_f(w_qkv, j0, kt * 32, NCH, lane);
    for (int mt = 0; mt < 8; ++mt) {
      v16h bx[4];                          // distinct regs: loads overlap,
#pragma unroll
      for (int kt = 0; kt < 4; ++kt)       // then 4 chained WMMAs
        bx[kt] = load_frag_h(xs, mt * 16, kt * 32, NCH, lane);
      v8f acc = {};
#pragma unroll
      for (int kt = 0; kt < 4; ++kt) acc = wmma_f16(aw[kt], bx[kt], acc);
      const int tok = mt * 16 + nl;        // D: n = token, m = j0 + 8g + r
      if (kind == 0) {
        h16* dst = qs + tok * NCH + j0 + 8 * g;       // contiguous -> b128
#pragma unroll
        for (int r = 0; r < 8; ++r) dst[r] = (h16)(acc[r] * QSCALE);
      } else if (kind == 1) {
        h16* dst = ks + tok * NCH + (j0 - 128) + 8 * g;
#pragma unroll
        for (int r = 0; r < 8; ++r) dst[r] = (h16)acc[r];
      } else {                             // V^T [c][tok]: strided b16 pairs
#pragma unroll
        for (int r = 0; r < 8; ++r)
          vt[(j0 - 256 + 8 * g + r) * NTOK + tok] = (h16)acc[r];
      }
    }
  }
  __syncthreads();

  // ---- Preload this wave's Q fragments (B operand of S^T), retire qs -----
  v16h aqv[4];
#pragma unroll
  for (int pair = 0; pair < 4; ++pair) {
    const int idx = wave * 4 + pair;
    aqv[pair] = load_frag_h(qs, (idx & 7) * 16, (idx >> 3) * HD, NCH, lane);
  }
  __syncthreads();            // qs dead block-wide -> reuse as attn-out plane
  h16* os = qs;

  // ---- Attention, fully in registers: S^T = K Q^T (+mask as C), softmax,
  //      O^T = V^T P^T.  Lane owns half a softmax row -> 1 shuffle/reduce,
  //      and S^T D-layout re-fragments in-lane into the P B-operand.
#pragma unroll
  for (int pair = 0; pair < 4; ++pair) {
    const int idx  = wave * 4 + pair;      // SGPR
    const int q0   = (idx & 7) * 16;
    const int c0   = (idx >> 3) * HD;

    const float* mq = mask + (size_t)win * NTOK * NTOK + (q0 + nl) * NTOK;
    __builtin_prefetch(mq + g * 64, 0, 1);          // global_prefetch_b8

    v8f s[8];                                        // S^T(key=16mt+8g+r, q=q0+nl)
#pragma unroll
    for (int mt = 0; mt < 8; ++mt) {
      const float4 m0 = *reinterpret_cast<const float4*>(mq + mt * 16 + 8 * g);
      const float4 m1 = *reinterpret_cast<const float4*>(mq + mt * 16 + 8 * g + 4);
      v8f cm;                                        // mask folded in as C operand
      cm[0] = m0.x; cm[1] = m0.y; cm[2] = m0.z; cm[3] = m0.w;
      cm[4] = m1.x; cm[5] = m1.y; cm[6] = m1.z; cm[7] = m1.w;
      s[mt] = wmma_f16(load_frag_h(ks, mt * 16, c0, NCH, lane), aqv[pair], cm);
    }

    // softmax over keys: 64 values in-lane + partner half-row via xor 16
    float mx = -3.0e38f;
#pragma unroll
    for (int mt = 0; mt < 8; ++mt)
#pragma unroll
      for (int r = 0; r < 8; ++r) mx = fmaxf(mx, s[mt][r]);
    mx = fmaxf(mx, __shfl_xor(mx, 16, 32));
    float sum = 0.f;
#pragma unroll
    for (int mt = 0; mt < 8; ++mt)
#pragma unroll
      for (int r = 0; r < 8; ++r) {
        const float e = __expf(s[mt][r] - mx);
        s[mt][r] = e;
        sum += e;
      }
    sum += __shfl_xor(sum, 16, 32);
    const float inv = 1.0f / sum;

    // In-lane re-fragmentation: P B-frag half e needs key 32kt + kmap(e,g),
    // which is s[2kt][e] (e<8) / s[2kt+1][e-8] (e>=8) in this very lane.
    v16h pf[4];
#pragma unroll
    for (int kt = 0; kt < 4; ++kt)
#pragma unroll
      for (int e = 0; e < 8; ++e) {
        pf[kt][e]     = (h16)(s[2 * kt][e]     * inv);
        pf[kt][e + 8] = (h16)(s[2 * kt + 1][e] * inv);
      }

    // O^T(d, q) = sum_key V^T(d,key) P^T(key,q): A from vt, B = pf
#pragma unroll
    for (int dt = 0; dt < 2; ++dt) {
      v16h av[4];
#pragma unroll
      for (int kt = 0; kt < 4; ++kt)
        av[kt] = load_frag_h(vt, c0 + dt * 16, kt * 32, NTOK, lane);
      v8f o = {};
#pragma unroll
      for (int kt = 0; kt < 4; ++kt) o = wmma_f16(av[kt], pf[kt], o);
      h16* dst = os + (q0 + nl) * NCH + c0 + dt * 16 + 8 * g;  // contiguous
#pragma unroll
      for (int r = 0; r < 8; ++r) dst[r] = (h16)o[r];
    }
  }
  __syncthreads();

  // ---- Output projection^T: A = w_proj rows j, B = os^T ------------------
  // D: m = j, n = token -> each lane stores 8 contiguous floats (2x b128).
  {
    const int mt = wave;                   // token tile (SGPR)
    float* ob = out + (size_t)b * NTOK * NCH;
    v16h bos[4];
#pragma unroll
    for (int kt = 0; kt < 4; ++kt)
      bos[kt] = load_frag_h(os, mt * 16, kt * 32, NCH, lane);
    for (int nt = 0; nt < 8; ++nt) {
      v8f acc = {};
#pragma unroll
      for (int kt = 0; kt < 4; ++kt)
        acc = wmma_f16(load_frag_f(w_proj, nt * 16, kt * 32, NCH, lane),
                       bos[kt], acc);
      const float4 b0 = *reinterpret_cast<const float4*>(b_proj + nt * 16 + 8 * g);
      const float4 b1 = *reinterpret_cast<const float4*>(b_proj + nt * 16 + 8 * g + 4);
      float* dst = ob + (size_t)(mt * 16 + nl) * NCH + nt * 16 + 8 * g;
      float4 o0, o1;
      o0.x = acc[0] + b0.x; o0.y = acc[1] + b0.y;
      o0.z = acc[2] + b0.z; o0.w = acc[3] + b0.w;
      o1.x = acc[4] + b1.x; o1.y = acc[5] + b1.y;
      o1.z = acc[6] + b1.z; o1.w = acc[7] + b1.w;
      *reinterpret_cast<float4*>(dst)     = o0;     // 2x global_store_b128
      *reinterpret_cast<float4*>(dst + 4) = o1;
    }
  }
}

extern "C" void kernel_launch(void* const* d_in, const int* in_sizes, int n_in,
                              void* d_out, int out_size, void* d_ws, size_t ws_size,
                              hipStream_t stream) {
  (void)n_in; (void)out_size; (void)d_ws; (void)ws_size;
  const float* x      = (const float*)d_in[0];
  const float* mask   = (const float*)d_in[1];
  const float* w_qkv  = (const float*)d_in[2];
  const float* w_proj = (const float*)d_in[3];
  const float* b_proj = (const float*)d_in[4];
  float* out = (float*)d_out;

  const int nwin = in_sizes[0] / (NTOK * NCH);   // 4096
  hipLaunchKernelGGL(swin_window_attention_kernel,
                     dim3(nwin), dim3(256), 0, stream,
                     x, mask, w_qkv, w_proj, b_proj, out);
}
